// InputStream_70609262346901
// MI455X (gfx1250) — matrix operand order repaired and checked by reading, not compiled, not run on hardware
//
#include <hip/hip_runtime.h>
#include <stdint.h>

typedef __attribute__((ext_vector_type(16))) _Float16 v16h;
typedef __attribute__((ext_vector_type(8)))  float    v8f;
typedef __attribute__((ext_vector_type(4)))  int      v4i;
typedef __attribute__((ext_vector_type(2)))  int      v2i;

#define BB    16
#define TT    5
#define HH    256
#define WW    256
#define CIN   3
#define FF    8
#define NGATE 32    // 4*F
#define KPAD  128   // k-map: x -> [0,36) (3 ch + pad per tap), pad [36,40), h -> [40,112), pad [112,128)
#define KH    40    // h base in k-map (40 halves = 80 B, 16B aligned)

// tile: 16 wide x 8 tall = 128 pixels per block; halo 18 x 10
#define TLW   16
#define TLH   8
#define NPIXT (TLW*TLH)

#define LOG2E 1.44269504088896340736f

union Frag16 { v16h v; v4i q[2]; };

// Hardware transcendentals: v_exp_f32 / v_rcp_f32 (TRANS pipe)
__device__ __forceinline__ float fast_sigmoid(float z) {
  float e = __builtin_amdgcn_exp2f(-LOG2E * z);
  return __builtin_amdgcn_rcpf(1.f + e);
}
__device__ __forceinline__ float fast_tanh(float z) {
  float e = __builtin_amdgcn_exp2f(-2.f * LOG2E * z);
  return 2.f * __builtin_amdgcn_rcpf(1.f + e) - 1.f;
}

// ---- one-shot: build transposed fused weight matrix Wt[n][k] (f16) in global ws ----
__global__ __launch_bounds__(256)
void prep_weights_kernel(const float* __restrict__ wk,   // [3,3,Cin,32]
                         const float* __restrict__ wr,   // [3,3,F,32]
                         _Float16* __restrict__ wtG) {   // [32][128]
  for (int i = threadIdx.x; i < NGATE * KPAD; i += 256) {
    int n = i >> 7, k = i & 127;
    float wv = 0.f;
    if (k < 36) {
      int tap = k >> 2, c = k & 3;
      if (tap < 9 && c < 3) wv = wk[(tap * 3 + c) * 32 + n];
    } else if (k >= KH && k < KH + 72) {
      wv = wr[(k - KH) * 32 + n];
    }
    wtG[i] = (_Float16)wv;
  }
}

__global__ __launch_bounds__(256)
void convlstm_step_kernel(const float* __restrict__ x,      // [B,T,H,W,Cin]
                          const _Float16* __restrict__ wtG, // [32][128] fused W^T
                          const float* __restrict__ bias,   // [32]
                          const _Float16* __restrict__ h_prev, // [B,H,W,F]
                          _Float16* __restrict__ h_next,       // [B,H,W,F]
                          float* __restrict__ c_state,         // [B,H,W,F]
                          int t) {
  __shared__ __align__(16) _Float16 sP[NPIXT * KPAD];     // im2col patch, 32 KB
  __shared__ __align__(16) _Float16 sx[10 * 18 * 4];      // x halo [iy][ix][c pad4]
  __shared__ __align__(16) _Float16 sh[10 * 18 * FF];     // h halo, 16 B per pixel
  __shared__ float sz[NPIXT * NGATE];                     // pre-activations, 16 KB

  const v4i vzero4 = {0, 0, 0, 0};
  const v2i vzero2 = {0, 0};

  const int tid  = threadIdx.x;
  const int lane = tid & 31;
  const int wave = tid >> 5;

  // block -> (b, tile_y, tile_x): 16 x-tiles, 32 y-tiles, 16 images
  const int blk = blockIdx.x;
  const int tx  = blk & 15;
  const int ty  = (blk >> 4) & 31;
  const int b   = blk >> 9;
  const int x0  = tx * TLW;
  const int y0  = ty * TLH;

  // ---- stage halos: 10 x 18 halo pixels ----
  for (int i = tid; i < 10 * 18; i += 256) {
    int iy = i / 18, ix = i - iy * 18;
    int gy = y0 + iy - 1, gx = x0 + ix - 1;
    bool inb = (gy >= 0 && gy < HH && gx >= 0 && gx < WW);

    // x halo: 3 f32 -> f16 (+ zero pad channel)
    float v0 = 0.f, v1 = 0.f, v2 = 0.f;
    if (inb) {
      const float* gp = x + ((((size_t)b * TT + t) * HH + gy) * WW + gx) * CIN;
      v0 = gp[0]; v1 = gp[1]; v2 = gp[2];
    }
    int sxi = i * 4;
    sx[sxi + 0] = (_Float16)v0;
    sx[sxi + 1] = (_Float16)v1;
    sx[sxi + 2] = (_Float16)v2;
    sx[sxi + 3] = (_Float16)0.f;

    // h halo: one 16-byte chunk per pixel; async copy global->LDS when valid
    int shSlot = i * FF;
    if (inb && t != 0) {
      unsigned lds = (unsigned)(uintptr_t)(&sh[shSlot]);
      unsigned long long ga =
          (unsigned long long)(uintptr_t)(h_prev + (((size_t)b * HH + gy) * WW + gx) * FF);
      asm volatile("global_load_async_to_lds_b128 %0, %1, off"
                   :: "v"(lds), "v"(ga) : "memory");
    } else {
      *(v4i*)&sh[shSlot] = vzero4;   // zero: out-of-bounds or t==0
    }
  }
  // ---- zero only the pad columns of the patch ----
  for (int i = tid; i < NPIXT; i += 256) {
    _Float16* row = sP + i * KPAD;
    *(v2i*)(row + 36)  = vzero2;    // k 36..39
    *(v4i*)(row + 112) = vzero4;    // k 112..119
    *(v4i*)(row + 120) = vzero4;    // k 120..127
  }
  asm volatile("s_wait_asynccnt 0x0" ::: "memory");
  __syncthreads();

  // ---- pack im2col patch ----
  // h taps: 9 taps * 128 pixels, one aligned 16 B chunk each
  for (int i = tid; i < 9 * NPIXT; i += 256) {
    int tap = i >> 7, pix = i & (NPIXT - 1);
    int dy = tap / 3, dx = tap - dy * 3;
    int py = pix >> 4, px = pix & 15;
    *(v4i*)&sP[pix * KPAD + KH + tap * 8] =
        *(const v4i*)&sh[((py + dy) * 18 + (px + dx)) * FF];
  }
  // x taps: 9 taps * 128 pixels, one aligned 8 B chunk (3 ch + pad)
  for (int i = tid; i < 9 * NPIXT; i += 256) {
    int tap = i >> 7, pix = i & (NPIXT - 1);
    int dy = tap / 3, dx = tap - dy * 3;
    int py = pix >> 4, px = pix & 15;
    *(v2i*)&sP[pix * KPAD + tap * 4] =
        *(const v2i*)&sx[((py + dy) * 18 + (px + dx)) * 4];
  }
  __syncthreads();

  // ---- WMMA: 8 waves x (1 M-group x 2 N-groups); K = 4 steps of 32 ----
  const int n0 = lane & 15;
  const int koffA = (lane < 16) ? 0 : 8;    // A: halves 0-7 -> K koff+j, 8-15 -> K koff+16+j
  const int koffB = (lane < 16) ? 0 : 8;    // B: halves 0-7 -> K koff+j, 8-15 -> K koff+16+j
  const int mg = wave;                       // pixel row within tile
  const int pix = mg * 16 + n0;
  const _Float16* aBase = sP + pix * KPAD + koffA;

  // A fragments: load once, reuse across both N-groups
  v4i aq[4][2];
#pragma unroll
  for (int kk = 0; kk < 4; ++kk) {
    aq[kk][0] = *(const v4i*)(aBase + kk * 32);        // K = kk*32+koffA+0..7
    aq[kk][1] = *(const v4i*)(aBase + kk * 32 + 16);   // K = kk*32+koffA+16..23
  }

  const int mbase = mg * 16 + ((lane >= 16) ? 8 : 0);
#pragma unroll
  for (int ng = 0; ng < 2; ++ng) {
    const _Float16* bBase = wtG + (ng * 16 + n0) * KPAD + koffB;  // global (L2-resident)
    v8f acc = {};
#pragma unroll
    for (int kk = 0; kk < 4; ++kk) {
      Frag16 a, bf;
      a.q[0]  = aq[kk][0];
      a.q[1]  = aq[kk][1];
      bf.q[0] = *(const v4i*)(bBase + kk * 32);        // K = kk*32+koffB+0..7
      bf.q[1] = *(const v4i*)(bBase + kk * 32 + 16);   // K = kk*32+koffB+16..23
      acc = __builtin_amdgcn_wmma_f32_16x16x32_f16(
          /*neg_a=*/false, a.v, /*neg_b=*/false, bf.v,
          /*c_mod=*/(short)0, acc, /*reuse_a=*/false, /*reuse_b=*/false);
    }
    const int ncol = ng * 16 + n0;
#pragma unroll
    for (int r = 0; r < 8; ++r)
      sz[(mbase + r) * NGATE + ncol] = acc[r];
  }
  __syncthreads();

  // ---- gate math: 128 pixels x 8 channels (hardware TRANS ops) ----
  for (int i = tid; i < NPIXT * FF; i += 256) {
    int f = i & 7, p = i >> 3;
    int ppy = p >> 4, ppx = p & 15;
    float zi = sz[p * NGATE + f]      + bias[f];
    float zf = sz[p * NGATE + 8 + f]  + bias[8 + f];
    float zg = sz[p * NGATE + 16 + f] + bias[16 + f];
    float zo = sz[p * NGATE + 24 + f] + bias[24 + f];
    float gi = fast_sigmoid(zi);
    float gf = fast_sigmoid(zf);
    float gg = fast_tanh(zg);
    float go = fast_sigmoid(zo);
    int gy = y0 + ppy, gx = x0 + ppx;
    size_t idx = (((size_t)b * HH + gy) * WW + gx) * FF + f;
    float cold = (t == 0) ? 0.f : c_state[idx];
    float cnew = gf * cold + gi * gg;
    float hnew = go * fast_tanh(cnew);
    c_state[idx] = cnew;
    h_next[idx]  = (_Float16)hnew;
  }
}

__global__ __launch_bounds__(256)
void head_kernel(const _Float16* __restrict__ h,   // [B,H,W,F]
                 const float* __restrict__ gamma, const float* __restrict__ beta,
                 const float* __restrict__ mean,  const float* __restrict__ var,
                 const float* __restrict__ dw,    // [F,2]
                 const float* __restrict__ db,    // [2]
                 float* __restrict__ out) {       // [B,H,W,2]
  int idx = blockIdx.x * 256 + threadIdx.x;       // pixel id
  const int NPIX = BB * HH * WW;
  if (idx >= NPIX) return;
  float a0 = db[0], a1 = db[1];
#pragma unroll
  for (int f = 0; f < FF; ++f) {
    float xv = (float)h[(size_t)idx * FF + f];
    xv = (xv - mean[f]) * __builtin_amdgcn_rsqf(var[f] + 1e-3f) * gamma[f] + beta[f];
    xv = (xv >= 0.f) ? xv : 0.3f * xv;
    a0 += xv * dw[f * 2 + 0];
    a1 += xv * dw[f * 2 + 1];
  }
  out[(size_t)idx * 2 + 0] = a0;
  out[(size_t)idx * 2 + 1] = a1;
}

extern "C" void kernel_launch(void* const* d_in, const int* in_sizes, int n_in,
                              void* d_out, int out_size, void* d_ws, size_t ws_size,
                              hipStream_t stream) {
  const float* x     = (const float*)d_in[0];  // [16,5,256,256,3]
  const float* wk    = (const float*)d_in[1];  // [3,3,3,32]
  const float* wr    = (const float*)d_in[2];  // [3,3,8,32]
  const float* bias  = (const float*)d_in[3];  // [32]
  const float* gamma = (const float*)d_in[4];  // [8]
  const float* beta  = (const float*)d_in[5];  // [8]
  const float* mean  = (const float*)d_in[6];  // [8]
  const float* var   = (const float*)d_in[7];  // [8]
  const float* dw    = (const float*)d_in[8];  // [8,2]
  const float* db    = (const float*)d_in[9];  // [2]

  // workspace layout: hA (16 MB f16), hB (16 MB f16), c (33.5 MB f32), Wt (8 KB f16)
  char* ws = (char*)d_ws;
  const size_t NSTATE = (size_t)BB * HH * WW * FF;  // 8,388,608
  _Float16* hA  = (_Float16*)ws;
  _Float16* hB  = (_Float16*)(ws + NSTATE * sizeof(_Float16));
  float*    cS  = (float*)   (ws + 2 * NSTATE * sizeof(_Float16));
  _Float16* wtG = (_Float16*)(ws + 2 * NSTATE * sizeof(_Float16) + NSTATE * sizeof(float));

  prep_weights_kernel<<<1, 256, 0, stream>>>(wk, wr, wtG);

  const int nblocks = BB * (HH / TLH) * (WW / TLW);   // 8192
  for (int t = 0; t < TT; ++t) {
    const _Float16* hp = (t & 1) ? hB : hA;
    _Float16*       hn = (t & 1) ? hA : hB;
    convlstm_step_kernel<<<nblocks, 256, 0, stream>>>(x, wtG, bias, hp, hn, cS, t);
  }
  // T=5: final h written by t=4 lands in hB
  const int NPIX = BB * HH * WW;
  head_kernel<<<(NPIX + 255) / 256, 256, 0, stream>>>(hB, gamma, beta, mean, var,
                                                      dw, db, (float*)d_out);
}
